// TransformerBlock_81784767250432
// MI455X (gfx1250) — matrix operand order repaired and checked by reading, not compiled, not run on hardware
//
#include <hip/hip_runtime.h>

typedef unsigned short u16;
typedef unsigned int   u32;
typedef unsigned long long u64;

typedef __attribute__((ext_vector_type(16))) __bf16 v16bf;
typedef __attribute__((ext_vector_type(8)))  float  v8f;
typedef __attribute__((ext_vector_type(4)))  unsigned int v4u;
typedef __attribute__((ext_vector_type(8)))  int v8i;
typedef __attribute__((ext_vector_type(4)))  int v4i;

#define S_LEN  1024
#define BATCH  2
#define DMODEL 1024
#define NHEAD  16
#define KDIM   64
#define TLEN   2048
#define MEMS   1024   /* MEM * S */
#define MFFN   4096
#define HK     1024   /* NHEAD*KDIM */

__device__ __forceinline__ u16 f2bf(float f) {
  union { float f; u32 u; } x; x.f = f;
  u32 r = x.u + 0x7FFFu + ((x.u >> 16) & 1u);
  return (u16)(r >> 16);
}
__device__ __forceinline__ float bf2f(u16 h) {
  union { u32 u; float f; } x; x.u = ((u32)h) << 16;
  return x.f;
}

union Frag { v16bf v; u16 u[16]; uint4 q[2]; };

// ---------------------------------------------------------------- conversion
__global__ __launch_bounds__(256) void cvt_f32_to_bf16(const float* __restrict__ in,
                                                       u16* __restrict__ out, int n) {
  int i = blockIdx.x * 256 + threadIdx.x;
  if (i < n) out[i] = f2bf(in[i]);
}

// ------------------------------------------------- GEMM  C[M,N] = A[M,Kd] * W[N,Kd]^T
// flags: 1 = add bias (f32), 2 = relu, 4 = store bf16 (else f32)
// grid = (M/128, N/64), block = 256 (8 waves); wave -> 16x64 output tile
__global__ __launch_bounds__(256) void gemm_bf16_wmma(const u16* __restrict__ A,
                                                      const u16* __restrict__ W,
                                                      const float* __restrict__ bias,
                                                      void* __restrict__ out,
                                                      int Kd, int N, int flags) {
  const int lane = threadIdx.x & 31;
  const int wave = threadIdx.x >> 5;
  const int r    = lane & 15;
  const int half = lane >> 4;
  const int m0 = (blockIdx.x * 8 + wave) * 16;
  const int n0 = blockIdx.y * 64;

  const u16* arow = A + (size_t)(m0 + r) * Kd;
  const u16* wr0  = W + (size_t)(n0 +  0 + r) * Kd;
  const u16* wr1  = W + (size_t)(n0 + 16 + r) * Kd;
  const u16* wr2  = W + (size_t)(n0 + 32 + r) * Kd;
  const u16* wr3  = W + (size_t)(n0 + 48 + r) * Kd;

  v8f acc[4] = {};
  for (int kk = 0; kk < Kd; kk += 32) {
    Frag a, b0, b1, b2, b3;
    a.q[0] = *(const uint4*)(arow + kk + half * 16);
    a.q[1] = *(const uint4*)(arow + kk + half * 16 + 8);
    const int bo = kk + half * 16;
    b0.q[0] = *(const uint4*)(wr0 + bo); b0.q[1] = *(const uint4*)(wr0 + bo + 8);
    b1.q[0] = *(const uint4*)(wr1 + bo); b1.q[1] = *(const uint4*)(wr1 + bo + 8);
    b2.q[0] = *(const uint4*)(wr2 + bo); b2.q[1] = *(const uint4*)(wr2 + bo + 8);
    b3.q[0] = *(const uint4*)(wr3 + bo); b3.q[1] = *(const uint4*)(wr3 + bo + 8);
    acc[0] = __builtin_amdgcn_wmma_f32_16x16x32_bf16(false, a.v, false, b0.v, (short)0, acc[0], false, false);
    acc[1] = __builtin_amdgcn_wmma_f32_16x16x32_bf16(false, a.v, false, b1.v, (short)0, acc[1], false, false);
    acc[2] = __builtin_amdgcn_wmma_f32_16x16x32_bf16(false, a.v, false, b2.v, (short)0, acc[2], false, false);
    acc[3] = __builtin_amdgcn_wmma_f32_16x16x32_bf16(false, a.v, false, b3.v, (short)0, acc[3], false, false);
  }
#pragma unroll
  for (int nt = 0; nt < 4; ++nt) {
    const int col = n0 + nt * 16 + r;
    const float bv = (flags & 1) ? bias[col] : 0.0f;
#pragma unroll
    for (int j = 0; j < 8; ++j) {
      float v = acc[nt][j] + bv;
      if (flags & 2) v = fmaxf(v, 0.0f);
      const size_t off = (size_t)(m0 + j + half * 8) * (size_t)N + col;
      if (flags & 4) ((u16*)out)[off] = f2bf(v);
      else           ((float*)out)[off] = v;
    }
  }
}

// ---------------------------------------------------------------- attention
// grid = (S/128, H, B), block = 256 (8 waves, 16 query rows each)
// K_e / K_r key blocks staged via Tensor Data Mover (tensor_load_to_lds);
// V staged (transposed) by VALU; P.V / scores on WMMA.
__global__ __launch_bounds__(256) void attn_wmma(const u16* __restrict__ q,
                                                 const u16* __restrict__ ke,
                                                 const u16* __restrict__ kr,
                                                 const u16* __restrict__ vv,
                                                 const float* __restrict__ u_bias,
                                                 const float* __restrict__ v_bias,
                                                 u16* __restrict__ av) {
  __shared__ u16 ke_lds[32 * 64];
  __shared__ u16 kr_lds[32 * 64];
  __shared__ u16 vT_lds[64 * 32];     // transposed: [dim][key]
  __shared__ u16 p_lds[8][512];       // per-wave 16x32 probs

  const int lane = threadIdx.x & 31;
  const int wave = threadIdx.x >> 5;
  const int r    = lane & 15;
  const int half = lane >> 4;
  const int h = blockIdx.y, b = blockIdx.z;
  const int s0 = blockIdx.x * 128 + wave * 16;

  // Load q rows, add u/v biases, pack two K-step A-fragments each.
  Frag qu[2], qv[2];
  {
    const u16* qrow = q + (size_t)((s0 + r) * BATCH + b) * HK + h * KDIM;
    const float* ub = u_bias + h * KDIM;
    const float* vb = v_bias + h * KDIM;
#pragma unroll
    for (int i = 0; i < 2; ++i) {
#pragma unroll
      for (int e = 0; e < 8; ++e) {
        const int k0 = i * 32 + half * 16 + e;
        const int k1 = i * 32 + half * 16 + 8 + e;
        const float q0 = bf2f(qrow[k0]);
        const float q1 = bf2f(qrow[k1]);
        qu[i].u[e]     = f2bf(q0 + ub[k0]);
        qu[i].u[8 + e] = f2bf(q1 + ub[k1]);
        qv[i].u[e]     = f2bf(q0 + vb[k0]);
        qv[i].u[8 + e] = f2bf(q1 + vb[k1]);
      }
    }
  }

  v8f accv[4] = {};
  float mrun[8], lrun[8];
#pragma unroll
  for (int j = 0; j < 8; ++j) { mrun[j] = -1e30f; lrun[j] = 0.0f; }

  const int nblk = (MEMS + blockIdx.x * 128 + 128) / 32;  // block-uniform causal bound
  const float scale = 0.125f;                             // 1/sqrt(64)

  // Static part of the TDM descriptor group1 (CDNA5 ISA):
  //  data_size=2B, tensor_dim0=64, tensor_dim1=4096, tile=64x32, dim0_stride=B*HK=2048.
  v8i g1;
  g1[0] = (int)0x00010000u;          // workgroup_mask=0, data_size=1 (2 bytes)
  g1[1] = (int)(64u << 16);          // tensor_dim0[15:0] in bits[31:16]
  g1[2] = (int)(0x1000u << 16);      // tensor_dim0[31:16]=0 | tensor_dim1[15:0]=4096
  g1[3] = (int)(64u << 16);          // tensor_dim1[31:16]=0 | tile_dim0=64
  g1[4] = 32;                        // tile_dim1=32, tile_dim2=0
  g1[5] = (int)(BATCH * HK);         // tensor_dim0_stride low32 = 2048
  g1[6] = 0;
  g1[7] = 0;
  const v4i gz4 = {};                // groups 2/3 unused (2-D tensor)
  const v8i gz8 = {};                // trailing group (clang-23 6-arg form)
  const u32 lds_ke = (u32)(size_t)ke_lds;
  const u32 lds_kr = (u32)(size_t)kr_lds;
  const size_t head_off = (size_t)b * HK + (size_t)h * KDIM;

  for (int tb = 0; tb < nblk; ++tb) {
    const int tbase = tb * 32;
    __syncthreads();   // previous tile fully consumed before overwrite

    if (wave == 0) {
      // TDM: DMA the 32x64 K_e and K_r tiles into LDS (row stride B*HK elems).
      const u64 ga_ke = (u64)(size_t)(ke + (size_t)tbase * BATCH * HK + head_off);
      const u64 ga_kr = (u64)(size_t)(kr + (size_t)tbase * BATCH * HK + head_off);
      v4u g0;
      g0[0] = 1u;                                    // count=1
      g0[1] = lds_ke;                                // lds_addr
      g0[2] = (u32)ga_ke;                            // global_addr lo
      g0[3] = (u32)(ga_ke >> 32) | 0x80000000u;      // global_addr hi | type=2
      __builtin_amdgcn_tensor_load_to_lds(g0, g1, gz4, gz4, gz8, 0);
      g0[1] = lds_kr;
      g0[2] = (u32)ga_kr;
      g0[3] = (u32)(ga_kr >> 32) | 0x80000000u;
      __builtin_amdgcn_tensor_load_to_lds(g0, g1, gz4, gz4, gz8, 0);
    }

    // All 256 threads stage V transposed into LDS (overlaps with the TDM).
    for (int idx = threadIdx.x; idx < 1024; idx += 256) {
      const int key = idx >> 5;
      const int dp  = (idx & 31) * 2;
      const size_t grow = (size_t)((tbase + key) * BATCH + b) * HK + h * KDIM + dp;
      const u32 e2 = *(const u32*)(vv + grow);
      vT_lds[dp * 32 + key]       = (u16)(e2 & 0xFFFFu);
      vT_lds[(dp + 1) * 32 + key] = (u16)(e2 >> 16);
    }

    if (wave == 0) __builtin_amdgcn_s_wait_tensorcnt(0);
    __syncthreads();

    // Scores: two 16-key tiles, AC + BD fused into one accumulator (4 wmma each).
    v8f sc[2];
#pragma unroll
    for (int tt = 0; tt < 2; ++tt) {
      v8f s = {};
#pragma unroll
      for (int i = 0; i < 2; ++i) {
        Frag bke, bkr;
        const u16* kep = ke_lds + (tt * 16 + r) * 64 + i * 32 + half * 16;
        const u16* krp = kr_lds + (tt * 16 + r) * 64 + i * 32 + half * 16;
        bke.q[0] = *(const uint4*)(kep); bke.q[1] = *(const uint4*)(kep + 8);
        bkr.q[0] = *(const uint4*)(krp); bkr.q[1] = *(const uint4*)(krp + 8);
        s = __builtin_amdgcn_wmma_f32_16x16x32_bf16(false, qu[i].v, false, bke.v, (short)0, s, false, false);
        s = __builtin_amdgcn_wmma_f32_16x16x32_bf16(false, qv[i].v, false, bkr.v, (short)0, s, false, false);
      }
      sc[tt] = s;
    }
    // Scale + Transformer-XL causal mask (t <= MEMS + s).
#pragma unroll
    for (int tt = 0; tt < 2; ++tt)
#pragma unroll
      for (int j = 0; j < 8; ++j) {
        const int t  = tbase + tt * 16 + r;
        const int sq = s0 + j + half * 8;
        sc[tt][j] = sc[tt][j] * scale + ((t <= MEMS + sq) ? 0.0f : -1e9f);
      }

    // Online softmax: row max / sum across the 16 lanes of each half.
    float mnew[8];
#pragma unroll
    for (int j = 0; j < 8; ++j) mnew[j] = fmaxf(mrun[j], fmaxf(sc[0][j], sc[1][j]));
#pragma unroll
    for (int off = 1; off < 16; off <<= 1)
#pragma unroll
      for (int j = 0; j < 8; ++j) mnew[j] = fmaxf(mnew[j], __shfl_xor(mnew[j], off, 32));

    float alpha[8], psum[8];
#pragma unroll
    for (int j = 0; j < 8; ++j) { alpha[j] = __expf(mrun[j] - mnew[j]); mrun[j] = mnew[j]; }
#pragma unroll
    for (int tt = 0; tt < 2; ++tt)
#pragma unroll
      for (int j = 0; j < 8; ++j) sc[tt][j] = __expf(sc[tt][j] - mnew[j]);
#pragma unroll
    for (int j = 0; j < 8; ++j) psum[j] = sc[0][j] + sc[1][j];
#pragma unroll
    for (int off = 1; off < 16; off <<= 1)
#pragma unroll
      for (int j = 0; j < 8; ++j) psum[j] += __shfl_xor(psum[j], off, 32);
#pragma unroll
    for (int j = 0; j < 8; ++j) lrun[j] = lrun[j] * alpha[j] + psum[j];
#pragma unroll
    for (int nt = 0; nt < 4; ++nt)
#pragma unroll
      for (int j = 0; j < 8; ++j) accv[nt][j] *= alpha[j];

    // Re-lay P into the A-fragment layout via per-wave LDS.
    u16* pl = p_lds[wave];
#pragma unroll
    for (int tt = 0; tt < 2; ++tt)
#pragma unroll
      for (int j = 0; j < 8; ++j)
        pl[(j + half * 8) * 32 + tt * 16 + r] = f2bf(sc[tt][j]);

    Frag ap;
    const u16* pr = pl + r * 32;
    ap.q[0] = *(const uint4*)(pr + half * 16);
    ap.q[1] = *(const uint4*)(pr + half * 16 + 8);

    // AV += P(16x32) x V(32x64): 4 wmma over head-dim tiles.
#pragma unroll
    for (int nt = 0; nt < 4; ++nt) {
      Frag bv;
      const u16* vp = vT_lds + (nt * 16 + r) * 32 + half * 16;
      bv.q[0] = *(const uint4*)(vp); bv.q[1] = *(const uint4*)(vp + 8);
      accv[nt] = __builtin_amdgcn_wmma_f32_16x16x32_bf16(false, ap.v, false, bv.v, (short)0, accv[nt], false, false);
    }
  }

  // Normalize and store AV (bf16) for the output projection GEMM.
#pragma unroll
  for (int nt = 0; nt < 4; ++nt)
#pragma unroll
    for (int j = 0; j < 8; ++j) {
      const int sq = s0 + j + half * 8;
      const float val = accv[nt][j] / lrun[j];
      av[(size_t)(sq * BATCH + b) * HK + h * KDIM + nt * 16 + r] = f2bf(val);
    }
}

// ---------------------------------------------------------------- layernorm (fused residual)
// grid = rows (S*B), block = 256; D = 1024 (4 elems / thread)
__global__ __launch_bounds__(256) void layernorm_res(const float* __restrict__ a,
                                                     const float* __restrict__ res,
                                                     const float* __restrict__ g,
                                                     const float* __restrict__ beta,
                                                     float* __restrict__ out_f32,
                                                     u16* __restrict__ out_bf16) {
  __shared__ float red[8];
  const int row  = blockIdx.x;
  const int lane = threadIdx.x & 31;
  const int wave = threadIdx.x >> 5;
  const float* ar = a   + (size_t)row * DMODEL;
  const float* rr = res + (size_t)row * DMODEL;

  float vals[4];
  float sum = 0.0f;
#pragma unroll
  for (int i = 0; i < 4; ++i) {
    const int c = threadIdx.x + i * 256;
    vals[i] = ar[c] + rr[c];
    sum += vals[i];
  }
#pragma unroll
  for (int off = 16; off > 0; off >>= 1) sum += __shfl_xor(sum, off, 32);
  if (lane == 0) red[wave] = sum;
  __syncthreads();
  float tot = 0.0f;
#pragma unroll
  for (int i = 0; i < 8; ++i) tot += red[i];
  const float mu = tot * (1.0f / DMODEL);
  __syncthreads();

  float sq = 0.0f;
#pragma unroll
  for (int i = 0; i < 4; ++i) { const float d = vals[i] - mu; sq += d * d; }
#pragma unroll
  for (int off = 16; off > 0; off >>= 1) sq += __shfl_xor(sq, off, 32);
  if (lane == 0) red[wave] = sq;
  __syncthreads();
  float var = 0.0f;
#pragma unroll
  for (int i = 0; i < 8; ++i) var += red[i];
  const float inv = rsqrtf(var * (1.0f / DMODEL) + 1e-5f);

#pragma unroll
  for (int i = 0; i < 4; ++i) {
    const int c = threadIdx.x + i * 256;
    const float y = (vals[i] - mu) * inv * g[c] + beta[c];
    out_f32[(size_t)row * DMODEL + c] = y;
    if (out_bf16) out_bf16[(size_t)row * DMODEL + c] = f2bf(y);
  }
}

// ---------------------------------------------------------------- host
extern "C" void kernel_launch(void* const* d_in, const int* in_sizes, int n_in,
                              void* d_out, int out_size, void* d_ws, size_t ws_size,
                              hipStream_t stream) {
  (void)in_sizes; (void)n_in; (void)out_size; (void)ws_size;
  const float* x_f    = (const float*)d_in[0];
  const float* p_f    = (const float*)d_in[1];
  /* d_in[2] = mask (computed analytically) */
  const float* mem_f  = (const float*)d_in[3];
  const float* ub_f   = (const float*)d_in[4];
  const float* vb_f   = (const float*)d_in[5];
  const float* wq_f   = (const float*)d_in[6];
  const float* wke_f  = (const float*)d_in[7];
  const float* wkr_f  = (const float*)d_in[8];
  const float* wv_f   = (const float*)d_in[9];
  const float* wc_f   = (const float*)d_in[10];
  const float* w1_f   = (const float*)d_in[11];
  const float* w1b_f  = (const float*)d_in[12];
  const float* w2_f   = (const float*)d_in[13];
  const float* w2b_f  = (const float*)d_in[14];
  const float* ln1g_f = (const float*)d_in[15];
  const float* ln1b_f = (const float*)d_in[16];
  const float* ln2g_f = (const float*)d_in[17];
  const float* ln2b_f = (const float*)d_in[18];

  char* ws = (char*)d_ws;
  size_t off = 0;
  auto alloc = [&](size_t bytes) { size_t c = off; off += (bytes + 255) & ~(size_t)255; return c; };

  const size_t n_xcat = (size_t)TLEN * BATCH * DMODEL;   // 4M
  const size_t n_rows = (size_t)S_LEN * BATCH;           // 2048
  const size_t n_act  = n_rows * DMODEL;                 // 2M

  const size_t o_xcat = alloc(n_xcat * 2);               // bf16 [T*B, D]  (also reused as z1)
  const size_t o_p    = alloc(n_xcat * 2);               // bf16 [T*B, D]
  const size_t o_wq   = alloc((size_t)HK * DMODEL * 2);
  const size_t o_wke  = alloc((size_t)HK * DMODEL * 2);
  const size_t o_wkr  = alloc((size_t)HK * DMODEL * 2);
  const size_t o_wv   = alloc((size_t)HK * DMODEL * 2);
  const size_t o_wc   = alloc((size_t)DMODEL * HK * 2);
  const size_t o_w1   = alloc((size_t)MFFN * DMODEL * 2);
  const size_t o_w2   = alloc((size_t)DMODEL * MFFN * 2);
  const size_t o_q    = alloc(n_act * 2);                // bf16 [S*B, HK]
  const size_t o_ke   = alloc(n_xcat * 2);               // bf16 [T*B, HK]
  const size_t o_kr   = alloc(n_xcat * 2);
  const size_t o_v    = alloc(n_xcat * 2);
  const size_t o_av   = alloc(n_act * 2);                // bf16
  const size_t o_u1   = alloc(n_act * 4);                // f32 (wc out)
  const size_t o_u    = alloc(n_act * 4);                // f32 (after LN1)
  const size_t o_ub   = alloc(n_act * 2);                // bf16 copy of u
  const size_t o_z2   = alloc(n_act * 4);                // f32 (FFN2 out)
  const size_t o_z1   = o_xcat;                          // bf16 [S*B, MFFN] reuses xcat+p (dead region)

  auto cvt = [&](const float* src, size_t dst, int n) {
    cvt_f32_to_bf16<<<dim3((n + 255) / 256), dim3(256), 0, stream>>>(src, (u16*)(ws + dst), n);
  };
  // Conversions: memory || x -> xcat, p, all weights.
  cvt(mem_f, o_xcat, (int)((size_t)MEMS * BATCH * DMODEL));
  cvt(x_f, o_xcat + (size_t)MEMS * BATCH * DMODEL * 2, (int)n_act);
  cvt(p_f,   o_p,   (int)n_xcat);
  cvt(wq_f,  o_wq,  HK * DMODEL);
  cvt(wke_f, o_wke, HK * DMODEL);
  cvt(wkr_f, o_wkr, HK * DMODEL);
  cvt(wv_f,  o_wv,  HK * DMODEL);
  cvt(wc_f,  o_wc,  DMODEL * HK);
  cvt(w1_f,  o_w1,  MFFN * DMODEL);
  cvt(w2_f,  o_w2,  DMODEL * MFFN);

  auto gemm = [&](size_t a, size_t w, const float* bias, size_t outo, int M, int N, int Kd, int flags) {
    gemm_bf16_wmma<<<dim3(M / 128, N / 64), dim3(256), 0, stream>>>(
        (const u16*)(ws + a), (const u16*)(ws + w), bias, (void*)(ws + outo), Kd, N, flags);
  };

  const size_t o_x_rows = o_xcat + (size_t)MEMS * BATCH * DMODEL * 2;  // bf16 x rows inside xcat
  gemm(o_x_rows, o_wq,  nullptr, o_q,  2048, HK, DMODEL, 4);   // Q   -> bf16
  gemm(o_xcat,   o_wke, nullptr, o_ke, 4096, HK, DMODEL, 4);   // K_e -> bf16
  gemm(o_p,      o_wkr, nullptr, o_kr, 4096, HK, DMODEL, 4);   // K_r -> bf16
  gemm(o_xcat,   o_wv,  nullptr, o_v,  4096, HK, DMODEL, 4);   // V   -> bf16

  attn_wmma<<<dim3(S_LEN / 128, NHEAD, BATCH), dim3(256), 0, stream>>>(
      (const u16*)(ws + o_q), (const u16*)(ws + o_ke), (const u16*)(ws + o_kr),
      (const u16*)(ws + o_v), ub_f, vb_f, (u16*)(ws + o_av));

  gemm(o_av, o_wc, nullptr, o_u1, 2048, DMODEL, HK, 0);        // output proj -> f32

  layernorm_res<<<dim3((int)n_rows), dim3(256), 0, stream>>>(
      (const float*)(ws + o_u1), x_f, ln1g_f, ln1b_f,
      (float*)(ws + o_u), (u16*)(ws + o_ub));

  gemm(o_ub, o_w1, w1b_f, o_z1, 2048, MFFN, DMODEL, 1 | 2 | 4); // FFN1: bias+relu -> bf16
  gemm(o_z1, o_w2, w2b_f, o_z2, 2048, DMODEL, MFFN, 1);         // FFN2: bias -> f32

  layernorm_res<<<dim3((int)n_rows), dim3(256), 0, stream>>>(
      (const float*)(ws + o_z2), (const float*)(ws + o_u), ln2g_f, ln2b_f,
      (float*)d_out, nullptr);
}
